// QLinear_30133490549674
// MI455X (gfx1250) — compile-verified, matrix operand order checked by reading, not run on hardware
//
#include <hip/hip_runtime.h>
#include <hip/hip_bf16.h>

#define BETA      0.05f
#define INV_BETA  20.0f
#define MDIM      4096
#define NDIM      4096
#define KDIM      4096

typedef _Float16 v8h  __attribute__((ext_vector_type(8)));
typedef _Float16 v16h __attribute__((ext_vector_type(16)));
typedef _Float16 v4h  __attribute__((ext_vector_type(4)));
typedef float    v8f  __attribute__((ext_vector_type(8)));

// ---------------------------------------------------------------------------
// Pass 1: D4 lattice quantize weight -> fp16 integer lattice coords (w/beta).
// One thread per 4-element group. round-half-even matches jnp.round.
// ---------------------------------------------------------------------------
__global__ __launch_bounds__(256) void d4_quantize_kernel(
    const float* __restrict__ w, _Float16* __restrict__ wq, int ngroups) {
  int g = blockIdx.x * blockDim.x + threadIdx.x;
  if (g >= ngroups) return;
  float4 wv = reinterpret_cast<const float4*>(w)[g];
  float v[4] = {wv.x * INV_BETA, wv.y * INV_BETA, wv.z * INV_BETA, wv.w * INV_BETA};
  float f[4], d[4];
  int s = 0;
#pragma unroll
  for (int i = 0; i < 4; ++i) {
    f[i] = rintf(v[i]);        // round half to even, like jnp.round
    d[i] = v[i] - f[i];
    s += (int)f[i];
  }
  if (s & 1) {                 // parity fix: flip worst coordinate
    int k = 0; float m = fabsf(d[0]);
#pragma unroll
    for (int i = 1; i < 4; ++i) {
      float ad = fabsf(d[i]);
      if (ad > m) { m = ad; k = i; }   // strict > keeps first max (jnp.argmax)
    }
    f[k] += (d[k] >= 0.0f) ? 1.0f : -1.0f;
  }
  v4h o = {(_Float16)f[0], (_Float16)f[1], (_Float16)f[2], (_Float16)f[3]};
  reinterpret_cast<v4h*>(wq)[g] = o;   // exact small integers in fp16
}

// ---------------------------------------------------------------------------
// Pass 2: x fp32 -> fp16 (bandwidth pass, float4 in / half4 out)
// ---------------------------------------------------------------------------
__global__ __launch_bounds__(256) void cvt_f16_kernel(
    const float* __restrict__ x, _Float16* __restrict__ xh, int n4) {
  int i = blockIdx.x * blockDim.x + threadIdx.x;
  if (i >= n4) return;
  float4 v = reinterpret_cast<const float4*>(x)[i];
  v4h o = {(_Float16)v.x, (_Float16)v.y, (_Float16)v.z, (_Float16)v.w};
  reinterpret_cast<v4h*>(xh)[i] = o;
}

// ---------------------------------------------------------------------------
// Pass 3: WMMA GEMM  out[M,N] = beta * (xh[M,K] @ wq[N,K]^T) + bias[N]
// 8 wave32 per block arranged 2(M) x 4(N) -> 128x256 block tile.
// Each wave: 64x64 tile = 4x4 v_wmma_f32_16x16x32_f16 accumulators, K-step 32.
// All 8 fragments hoisted into distinct registers -> one 16-load burst per
// K-step, 16 WMMAs drain behind staggered s_wait_loadcnt.
// Fragment row offsets are immediates (i*16*KDIM*2 B < 2^23) -> 2 base ptrs.
// ---------------------------------------------------------------------------
__global__ __launch_bounds__(256) void gemm_wmma_kernel(
    const _Float16* __restrict__ xh, const _Float16* __restrict__ wq,
    const float* __restrict__ bias, float* __restrict__ out) {
  const int lane = threadIdx.x & 31;
  const int wid  = threadIdx.x >> 5;   // 0..7
  const int l16  = lane & 15;
  const int hig  = lane >> 4;          // 0/1: which K-half this lane holds
  const int wm   = wid >> 2;           // 0..1
  const int wn   = wid & 3;            // 0..3

  const int m0 = blockIdx.y * 128 + wm * 64;
  const int n0 = blockIdx.x * 256 + wn * 64;

  const v8f zero = {0.f, 0.f, 0.f, 0.f, 0.f, 0.f, 0.f, 0.f};
  v8f acc[4][4];
#pragma unroll
  for (int i = 0; i < 4; ++i)
#pragma unroll
    for (int j = 0; j < 4; ++j) acc[i][j] = zero;

  // Two per-lane base pointers; fragment rows addressed by immediate offsets.
  // (16B aligned: row*K*2 is 8KB-aligned, +hig*16B)
  const _Float16* ap = xh + (size_t)(m0 + l16) * KDIM + hig * 8;
  const _Float16* bp = wq + (size_t)(n0 + l16) * KDIM + hig * 8;

  for (int k = 0; k < KDIM; k += 32) {
    if (k + 256 < KDIM) {   // prefetch ~8 K-steps ahead (global_prefetch_b8)
#pragma unroll
      for (int i = 0; i < 4; ++i)
        __builtin_prefetch(ap + i * 16 * KDIM + k + 256, 0, 3);
#pragma unroll
      for (int j = 0; j < 4; ++j)
        __builtin_prefetch(bp + j * 16 * KDIM + k + 256, 0, 3);
    }
    v16h a[4], b[4];
#pragma unroll
    for (int i = 0; i < 4; ++i) {
      const _Float16* p = ap + i * 16 * KDIM + k;
      v8h lo = *reinterpret_cast<const v8h*>(p);        // K = k+hig*8 ..
      v8h hi = *reinterpret_cast<const v8h*>(p + 16);   // K = k+16+hig*8 ..
      a[i] = __builtin_shufflevector(lo, hi, 0,1,2,3,4,5,6,7,8,9,10,11,12,13,14,15);
    }
#pragma unroll
    for (int j = 0; j < 4; ++j) {
      const _Float16* p = bp + j * 16 * KDIM + k;
      v8h lo = *reinterpret_cast<const v8h*>(p);
      v8h hi = *reinterpret_cast<const v8h*>(p + 16);
      b[j] = __builtin_shufflevector(lo, hi, 0,1,2,3,4,5,6,7,8,9,10,11,12,13,14,15);
    }
#pragma unroll
    for (int j = 0; j < 4; ++j)
#pragma unroll
      for (int i = 0; i < 4; ++i)
        acc[i][j] = __builtin_amdgcn_wmma_f32_16x16x32_f16(
            /*neg_a=*/false, a[i], /*neg_b=*/false, b[j],
            /*c_mod=*/(short)0, acc[i][j], /*reuse_a=*/false, /*reuse_b=*/false);
  }

  // Epilogue: D layout — VGPR r: lanes 0-15 -> M=r, lanes 16-31 -> M=8+r; N=l16
  float bj[4];
#pragma unroll
  for (int j = 0; j < 4; ++j) bj[j] = bias[n0 + j * 16 + l16];
#pragma unroll
  for (int i = 0; i < 4; ++i)
#pragma unroll
    for (int r = 0; r < 8; ++r) {
      const size_t rowbase = (size_t)(m0 + i * 16 + hig * 8 + r) * NDIM;
#pragma unroll
      for (int j = 0; j < 4; ++j)
        out[rowbase + n0 + j * 16 + l16] = acc[i][j][r] * BETA + bj[j];
    }
}

// ---------------------------------------------------------------------------
extern "C" void kernel_launch(void* const* d_in, const int* in_sizes, int n_in,
                              void* d_out, int out_size, void* d_ws, size_t ws_size,
                              hipStream_t stream) {
  const float* x      = (const float*)d_in[0];   // [4096, 4096]
  const float* weight = (const float*)d_in[1];   // [4096, 4096]
  const float* bias   = (const float*)d_in[2];   // [4096]
  float*       out    = (float*)d_out;           // [4096, 4096]

  // workspace: 32MB xh + 32MB wq (fp16)
  _Float16* xh = (_Float16*)d_ws;
  _Float16* wq = xh + (size_t)MDIM * KDIM;

  {
    int n4 = (MDIM * KDIM) / 4;
    cvt_f16_kernel<<<(n4 + 255) / 256, 256, 0, stream>>>(x, xh, n4);
  }
  {
    int ng = (NDIM * KDIM) / 4;
    d4_quantize_kernel<<<(ng + 255) / 256, 256, 0, stream>>>(weight, wq, ng);
  }
  {
    dim3 grid(NDIM / 256, MDIM / 128);
    gemm_wmma_kernel<<<grid, 256, 0, stream>>>(xh, wq, bias, out);
  }
}
